// LstmModel_9431748182398
// MI455X (gfx1250) — matrix operand order repaired and checked by reading, not compiled, not run on hardware
//
#include <hip/hip_runtime.h>
#include <cstdint>
#include <cstddef>

typedef __attribute__((ext_vector_type(16))) _Float16 v16h;
typedef __attribute__((ext_vector_type(8)))  _Float16 v8h;
typedef __attribute__((ext_vector_type(8)))  float    v8f;

#define DEV __device__ __forceinline__

constexpr int Bn = 16, Rn = 64, Sn = 512, Cn = 64, Wn = 8, Tn = 384, Dn = 256, Hn = 4;
constexpr int Nn = Bn * Rn;
constexpr float EPSc = 1e-5f;

DEV int laneid() { return (int)(threadIdx.x & 31u); }

// ---- WMMA fragment helpers (per CDNA5 ISA 7.12.2 layouts) ----

// A fragment: 16x32 f16 tile, row-major, leading dim ldk (elements).
// Lane m(+16h): elements 0-7 = K 0..7 (+8h), elements 8-15 = K 16..23 (+8h)
// -> exactly two contiguous 16B LDS loads per lane.
DEV v16h ld_a(const _Float16* base, int ldk) {
  int m = laneid() & 15, h = laneid() >> 4;
  const _Float16* row = base + m * ldk + 8 * h;
  v8h lo = *(const v8h*)row;          // K 0..7  (+8h)
  v8h hi = *(const v8h*)(row + 16);   // K 16..23 (+8h)
  v16h a;
#pragma unroll
  for (int e = 0; e < 8; ++e) { a[e] = lo[e]; a[8 + e] = hi[e]; }
  return a;
}

// B fragment: 32x16, weight stored [N][K] row-major. lane n(+16h) holds column n,
// elements 0-7 = K 0..7 (+8h), elements 8-15 = K 16..23 (+8h) -> two 16B loads.
DEV v16h ld_b(const _Float16* W, int n0, int ldk, int k0) {
  const _Float16* p = W + (size_t)(n0 + (laneid() & 15)) * ldk + k0 + 8 * (laneid() >> 4);
  v8h lo = *(const v8h*)p;
  v8h hi = *(const v8h*)(p + 16);
  v16h b;
#pragma unroll
  for (int e = 0; e < 8; ++e) { b[e] = lo[e]; b[8 + e] = hi[e]; }
  return b;
}

DEV v8f wmma_f16(v16h a, v16h b, v8f c) {
  return __builtin_amdgcn_wmma_f32_16x16x32_f16(false, a, false, b, (short)0, c, false, false);
}

// ---------------- weight conversion kernels ----------------

__global__ void k_cvt(const float* __restrict__ s, _Float16* __restrict__ d, int n) {
  int i = blockIdx.x * 256 + threadIdx.x;
  if (i < n) d[i] = (_Float16)s[i];
}

__global__ void k_cvt_conv(const float* __restrict__ w, _Float16* __restrict__ d,
                           int Cin, int Kpad) {
  int i = blockIdx.x * 256 + threadIdx.x;
  if (i >= 64 * Kpad) return;
  int co = i / Kpad, k = i - co * Kpad;
  d[i] = (k < 3 * Cin) ? (_Float16)w[co * 3 * Cin + k] : (_Float16)0.f;
}

__global__ void k_cvt_lstm(const float* __restrict__ wih, const float* __restrict__ whh,
                           _Float16* __restrict__ d) {
  int i = blockIdx.x * 256 + threadIdx.x;
  if (i >= 1024 * 512) return;
  int n = i >> 9, k = i & 511;
  d[i] = (_Float16)(k < 256 ? wih[n * 256 + k] : whh[n * 256 + (k - 256)]);
}

__global__ void k_transpose(const float* __restrict__ reads, _Float16* __restrict__ xin) {
  int n = blockIdx.x;
  for (int idx = threadIdx.x; idx < 5 * 512; idx += 256) {
    int c = idx >> 9, s = idx & 511;
    xin[((size_t)n * 5 + c) * Sn + s] = (_Float16)reads[((size_t)n * Sn + s) * 5 + c];
  }
}

// ---------------- conv layer (im2col WMMA) ----------------
template <bool OUT32>
__global__ __launch_bounds__(256) void k_conv(
    const _Float16* __restrict__ xin, const _Float16* __restrict__ wf,
    const float* __restrict__ bias, void* __restrict__ outp, int Cin, int Kpad) {
  extern __shared__ char smem[];
  const int Kld = Kpad + 16;
  _Float16* patch = (_Float16*)smem;              // [128][Kld]
  _Float16* xs    = patch + 128 * Kld;            // [Cin][132]
  const int n = blockIdx.y, s0 = blockIdx.x * 128;
  for (int idx = threadIdx.x; idx < Cin * 130; idx += 256) {
    int ci = idx / 130, sl = idx - ci * 130;
    int s = s0 + sl - 1;
    _Float16 v = (_Float16)0.f;
    if (s >= 0 && s < Sn) v = xin[((size_t)n * Cin + ci) * Sn + s];
    xs[ci * 132 + sl] = v;
  }
  __syncthreads();
  const int Kvalid = 3 * Cin;
  for (int idx = threadIdx.x; idx < 128 * Kpad; idx += 256) {
    int k = idx >> 7, p = idx & 127;
    _Float16 v = (_Float16)0.f;
    if (k < Kvalid) { int ci = k / 3, t = k - 3 * ci; v = xs[ci * 132 + p + t]; }
    patch[p * Kld + k] = v;
  }
  __syncthreads();
  int wave = threadIdx.x >> 5, nl = laneid() & 15, hl = laneid() >> 4;
  v8f acc[4] = {};
  v16h bf = ld_b(wf, 0, Kpad, 0);                 // software-pipelined B
  for (int ks = 0; ks < Kpad / 32; ++ks) {
    v16h a = ld_a(patch + (wave * 16) * Kld + ks * 32, Kld);
#pragma unroll
    for (int ct = 0; ct < 4; ++ct) {
      v16h bn = ld_b(wf, (((ct + 1) & 3)) * 16, Kpad, (ks + (ct == 3)) * 32);
      acc[ct] = wmma_f16(a, bf, acc[ct]);
      bf = bn;
    }
  }
#pragma unroll
  for (int ct = 0; ct < 4; ++ct) {
    int co = ct * 16 + nl;
    float bb = bias[co];
#pragma unroll
    for (int r = 0; r < 8; ++r) {
      int pos = s0 + wave * 16 + r + 8 * hl;
      float v = fmaxf(acc[ct][r] + bb, 0.f);
      size_t o = ((size_t)n * Cn + co) * Sn + pos;
      if (OUT32) ((float*)outp)[o] = v;
      else       ((_Float16*)outp)[o] = (_Float16)v;
    }
  }
}

// ---------------- persistent scan kernel: one workgroup per batch ----------------
__global__ __launch_bounds__(512, 1) void k_scan(
    const _Float16* __restrict__ orig, float* __restrict__ belief,
    const _Float16* __restrict__ W1,  const float* __restrict__ B1,
    const _Float16* __restrict__ W2,  const float* __restrict__ B2,
    const _Float16* __restrict__ WL,  const float* __restrict__ BIH, const float* __restrict__ BHH,
    const _Float16* __restrict__ WQ,  const float* __restrict__ BQ,
    const _Float16* __restrict__ WK,  const float* __restrict__ BK,
    const _Float16* __restrict__ WV,  const float* __restrict__ BV,
    const _Float16* __restrict__ WG,  const float* __restrict__ BG,
    const _Float16* __restrict__ WU1, const float* __restrict__ BU1,
    const _Float16* __restrict__ WU2, const float* __restrict__ BU2,
    float* __restrict__ gfs) {
  extern __shared__ char smem[];
  _Float16* sWin   = (_Float16*)smem;            // 64*512 f16: win A halves / q|k / u1out
  _Float16* sAct   = sWin + 64 * 512;            // 64*512 f16: act1 / ucat
  _Float16* sXWH   = sAct + 64 * 512;            // 64*512 f16: [xw | hb] (hb persistent)
  float*    sC     = (float*)(sXWH + 64 * 512);  // 64*256 f32 cell state (persistent)
  _Float16* sH     = (_Float16*)(sC + 64 * 256); // 64*256 f16 raw h
  float*    sScore = (float*)(sH + 64 * 256);    // 64*64 f32
  float*    sAm    = sScore + 64 * 64;           // 64
  float*    sZ     = sAm + 64;                   // 256
  float*    sHd    = sZ + 256;                   // 1024
  float*    sGf    = sHd + 1024;                 // 256

  const int b = blockIdx.x, tid = threadIdx.x, wave = tid >> 5;
  const int nl = laneid() & 15, hl = laneid() >> 4;

  for (int i = tid; i < 64 * 256; i += 512) sC[i] = 0.f;
  for (int i = tid; i < 64 * 256; i += 512) {
    int r = i >> 8, d = i & 255;
    sXWH[r * 512 + 256 + d] = (_Float16)0.f;     // h0 = 0
  }
  __syncthreads();

  for (int t = 0; t < Tn; ++t) {
    // ---- window half 1: instance-normalized belief window (K cols 0..511) ----
    for (int idx = tid; idx < 64 * 64; idx += 512) {
      int r = idx >> 6, c = idx & 63;
      const float* bp = belief + (((size_t)b * 64 + r) * 64 + c) * Sn + t;
      float w[8], s = 0.f, s2 = 0.f;
#pragma unroll
      for (int j = 0; j < 8; ++j) { w[j] = bp[j]; s += w[j]; s2 += w[j] * w[j]; }
      float m = s * 0.125f;
      float inv = rsqrtf(s2 * 0.125f - m * m + EPSc);
#pragma unroll
      for (int j = 0; j < 8; ++j) sWin[r * 512 + c * 8 + j] = (_Float16)((w[j] - m) * inv);
    }
    __syncthreads();

    // ---- G1: win[64,1024] x W1^T[1024,512], K split, B software-pipelined ----
    v8f acc1[8];
#pragma unroll
    for (int q = 0; q < 8; ++q) acc1[q] = {};
    const int m0g = (wave >> 2) * 16, nbg = (wave & 3) * 8;
    v16h bf1 = ld_b(W1, nbg * 16, 1024, 0);
    for (int ks = 0; ks < 16; ++ks) {
      __builtin_prefetch(W1 + (size_t)(nbg * 16 + nl) * 1024 + ks * 32 + 64, 0, 0);
      v16h a = ld_a(sWin + m0g * 512 + ks * 32, 512);
#pragma unroll
      for (int q = 0; q < 8; ++q) {
        v16h bn = ld_b(W1, (nbg + ((q + 1) & 7)) * 16, 1024, (ks + (q == 7)) * 32);
        acc1[q] = wmma_f16(a, bf1, acc1[q]);
        bf1 = bn;
      }
    }
    __syncthreads();
    // ---- window half 2: original window (K cols 512..1023) ----
    for (int idx = tid; idx < 64 * 64; idx += 512) {
      int r = idx >> 6, c = idx & 63;
      const _Float16* op = orig + (((size_t)b * 64 + r) * 64 + c) * Sn + t;
#pragma unroll
      for (int j = 0; j < 8; ++j) sWin[r * 512 + c * 8 + j] = op[j];
    }
    __syncthreads();
    // bf1 already holds (tile nbg, k0=512): continue seamlessly.
    for (int ks = 0; ks < 16; ++ks) {
      v16h a = ld_a(sWin + m0g * 512 + ks * 32, 512);
#pragma unroll
      for (int q = 0; q < 8; ++q) {
        v16h bn = ld_b(W1, (nbg + ((q + 1) & 7)) * 16, 1024, 512 + (ks + (q == 7)) * 32);
        acc1[q] = wmma_f16(a, bf1, acc1[q]);
        bf1 = bn;
      }
    }
#pragma unroll
    for (int q = 0; q < 8; ++q) {
      int col = (nbg + q) * 16 + nl;
      float bbv = B1[col];
#pragma unroll
      for (int r = 0; r < 8; ++r)
        sAct[(m0g + r + 8 * hl) * 512 + col] = (_Float16)fmaxf(acc1[q][r] + bbv, 0.f);
    }
    __syncthreads();

    // ---- G2: xw = relu(act1 @ W2^T + b2) -> sXWH[:,0:256] ----
    {
      int id0 = wave * 4;
      int m0 = (id0 >> 4) * 16, nb = id0 & 15;
      v8f acc[4];
#pragma unroll
      for (int q = 0; q < 4; ++q) acc[q] = {};
      v16h bf = ld_b(W2, nb * 16, 512, 0);
      for (int ks = 0; ks < 16; ++ks) {
        v16h a = ld_a(sAct + m0 * 512 + ks * 32, 512);
#pragma unroll
        for (int q = 0; q < 4; ++q) {
          v16h bn = ld_b(W2, (nb + ((q + 1) & 3)) * 16, 512, (ks + (q == 3)) * 32);
          acc[q] = wmma_f16(a, bf, acc[q]);
          bf = bn;
        }
      }
#pragma unroll
      for (int q = 0; q < 4; ++q) {
        int col = (nb + q) * 16 + nl;
        float bbv = B2[col];
#pragma unroll
        for (int r = 0; r < 8; ++r)
          sXWH[(m0 + r + 8 * hl) * 512 + col] = (_Float16)fmaxf(acc[q][r] + bbv, 0.f);
      }
    }
    __syncthreads();

    // ---- LSTM: gates = [xw|h] @ [Wih|Whh]^T, i/f/g/o tiles fused per wave ----
    for (int task = 0; task < 4; ++task) {
      int id = wave * 4 + task;                     // 64 tasks (m_tile, j)
      int m0 = (id >> 4) * 16, j = id & 15;
      v8f acc[4];
#pragma unroll
      for (int g = 0; g < 4; ++g) acc[g] = {};
      v16h bf = ld_b(WL, j * 16, 512, 0);
      for (int ks = 0; ks < 16; ++ks) {
        v16h a = ld_a(sXWH + m0 * 512 + ks * 32, 512);
#pragma unroll
        for (int g = 0; g < 4; ++g) {
          v16h bn = ld_b(WL, ((g + 1) & 3) * 256 + j * 16, 512, (ks + (g == 3)) * 32);
          acc[g] = wmma_f16(a, bf, acc[g]);
          bf = bn;
        }
      }
      int d = j * 16 + nl;
#pragma unroll
      for (int r = 0; r < 8; ++r) {
        int row = m0 + r + 8 * hl;
        float gi = acc[0][r] + BIH[d] + BHH[d];
        float gf = acc[1][r] + BIH[256 + d] + BHH[256 + d];
        float gg = acc[2][r] + BIH[512 + d] + BHH[512 + d];
        float go = acc[3][r] + BIH[768 + d] + BHH[768 + d];
        float si = 1.f / (1.f + __expf(-gi));
        float sf = 1.f / (1.f + __expf(-gf));
        float so = 1.f / (1.f + __expf(-go));
        float cn = sf * sC[row * 256 + d] + si * tanhf(gg);
        sC[row * 256 + d] = cn;
        sH[row * 256 + d] = (_Float16)(so * tanhf(cn));
      }
    }
    __syncthreads();

    // ---- normalize-hidden over reads dim per channel -> sXWH[:,256:512] ----
    if (tid < 256) {
      int d = tid;
      float s = 0.f, s2 = 0.f;
      for (int r = 0; r < 64; ++r) { float v = (float)sH[r * 256 + d]; s += v; s2 += v * v; }
      float m = s * (1.f / 64.f);
      float inv = rsqrtf(s2 * (1.f / 64.f) - m * m + EPSc);
      for (int r = 0; r < 64; ++r)
        sXWH[r * 512 + 256 + d] = (_Float16)(((float)sH[r * 256 + d] - m) * inv);
    }
    __syncthreads();

    // ---- attention per head; mean_r(att@v) folded to (mean_r att)@hb@Wv^T ----
    _Float16* sQ = sWin;
    _Float16* sK = sWin + 64 * 256;
    for (int hh = 0; hh < Hn; ++hh) {
      {                                             // q (waves 0-7) / k (waves 8-15)
        bool isK = wave >= 8;
        const _Float16* Wm = (isK ? WK : WQ) + (size_t)hh * 256 * 256;
        const float* Bm = isK ? BK : BQ;
        _Float16* Out = isK ? sK : sQ;
        int w8 = wave & 7;
        int m0 = (w8 >> 1) * 16, nb = (w8 & 1) * 8;
        v8f acc[8];
#pragma unroll
        for (int q = 0; q < 8; ++q) acc[q] = {};
        v16h bf = ld_b(Wm, nb * 16, 256, 0);
        for (int ks = 0; ks < 8; ++ks) {
          v16h a = ld_a(sXWH + 256 + m0 * 512 + ks * 32, 512);
#pragma unroll
          for (int q = 0; q < 8; ++q) {
            v16h bn = ld_b(Wm, (nb + ((q + 1) & 7)) * 16, 256, (ks + (q == 7)) * 32);
            acc[q] = wmma_f16(a, bf, acc[q]);
            bf = bn;
          }
        }
#pragma unroll
        for (int q = 0; q < 8; ++q) {
          int col = (nb + q) * 16 + nl;
          float bbv = Bm[hh * 256 + col];
#pragma unroll
          for (int r = 0; r < 8; ++r)
            Out[(m0 + r + 8 * hl) * 256 + col] = (_Float16)(acc[q][r] + bbv);
        }
      }
      __syncthreads();
      {                                             // scores = q @ k^T, 1 tile/wave
        int m0 = (wave >> 2) * 16, n0 = (wave & 3) * 16;
        v8f acc = {};
        v16h bf = ld_b(sK, n0, 256, 0);
        for (int ks = 0; ks < 8; ++ks) {
          v16h bn = ld_b(sK, n0, 256, (ks + 1) * 32);   // last is discarded over-read in LDS
          acc = wmma_f16(ld_a(sQ + m0 * 256 + ks * 32, 256), bf, acc);
          bf = bn;
        }
#pragma unroll
        for (int r = 0; r < 8; ++r)
          sScore[(m0 + r + 8 * hl) * 64 + n0 + nl] = acc[r];
      }
      __syncthreads();
      if (tid < 64) {                               // row softmax
        float mx = -1e30f;
        for (int s = 0; s < 64; ++s) mx = fmaxf(mx, sScore[tid * 64 + s]);
        float sum = 0.f;
        for (int s = 0; s < 64; ++s) { float e = __expf(sScore[tid * 64 + s] - mx); sScore[tid * 64 + s] = e; sum += e; }
        float inv = 1.f / sum;
        for (int s = 0; s < 64; ++s) sScore[tid * 64 + s] *= inv;
      }
      __syncthreads();
      if (tid < 64) {                               // column mean of att
        float s = 0.f;
        for (int r = 0; r < 64; ++r) s += sScore[r * 64 + tid];
        sAm[tid] = s * (1.f / 64.f);
      }
      __syncthreads();
      if (tid < 256) {                              // z = amean @ hb
        float z = 0.f;
        for (int s = 0; s < 64; ++s) z += sAm[s] * (float)sXWH[s * 512 + 256 + tid];
        sZ[tid] = z;
      }
      __syncthreads();
      if (tid < 256) {                              // hd = relu(Wv z + bv)
        const _Float16* Wvp = WV + (size_t)hh * 256 * 256 + (size_t)tid * 256;
        float a = BV[hh * 256 + tid];
        for (int dd = 0; dd < 256; ++dd) a += (float)Wvp[dd] * sZ[dd];
        sHd[hh * 256 + tid] = fmaxf(a, 0.f);
      }
      __syncthreads();
    }

    // ---- global feature: gf = relu(Wg @ hd_flat + bg) ----
    if (tid < 256) {
      const _Float16* Wgp = WG + (size_t)tid * 1024;
      float a = BG[tid];
      for (int j = 0; j < 1024; ++j) a += (float)Wgp[j] * sHd[j];
      a = fmaxf(a, 0.f);
      sGf[tid] = a;
      gfs[((size_t)b * Tn + t) * 256 + tid] = a;
    }
    __syncthreads();

    // ---- ucat = [hb | gf] -> sAct ----
    for (int i = tid; i < 64 * 512; i += 512) {
      int r = i >> 9, col = i & 511;
      sAct[i] = (col < 256) ? sXWH[r * 512 + 256 + col] : (_Float16)sGf[col - 256];
    }
    __syncthreads();

    // ---- u1: relu(ucat @ Wu1^T + bu1) -> sWin ----
    {
      v8f acc[8];
#pragma unroll
      for (int q = 0; q < 8; ++q) acc[q] = {};
      v16h bf = ld_b(WU1, nbg * 16, 512, 0);
      for (int ks = 0; ks < 16; ++ks) {
        v16h a = ld_a(sAct + m0g * 512 + ks * 32, 512);
#pragma unroll
        for (int q = 0; q < 8; ++q) {
          v16h bn = ld_b(WU1, (nbg + ((q + 1) & 7)) * 16, 512, (ks + (q == 7)) * 32);
          acc[q] = wmma_f16(a, bf, acc[q]);
          bf = bn;
        }
      }
#pragma unroll
      for (int q = 0; q < 8; ++q) {
        int col = (nbg + q) * 16 + nl;
        float bbv = BU1[col];
#pragma unroll
        for (int r = 0; r < 8; ++r)
          sWin[(m0g + r + 8 * hl) * 512 + col] = (_Float16)fmaxf(acc[q][r] + bbv, 0.f);
      }
    }
    __syncthreads();

    // ---- u2 + residual belief update (direct global RMW from fragments) ----
    for (int tt2 = 0; tt2 < 7; ++tt2) {
      int id = wave + 16 * tt2;                     // 112 tiles
      int m0 = (id / 28) * 16, n0t = id % 28;
      v8f acc = {};
      v16h bf = ld_b(WU2, n0t * 16, 512, 0);
      for (int ks = 0; ks < 16; ++ks) {
        v16h bn = ld_b(WU2, n0t * 16, 512, (ks + 1) * 32);
        acc = wmma_f16(ld_a(sWin + m0 * 512 + ks * 32, 512), bf, acc);
        bf = bn;
      }
      int col = n0t * 16 + nl;
      float bbv = BU2[col];
      int cch = col / 7, wj = col - cch * 7;
#pragma unroll
      for (int r = 0; r < 8; ++r) {
        int row = m0 + r + 8 * hl;
        float u = fmaxf(acc[r] + bbv, 0.f);
        float* bp = belief + (((size_t)b * 64 + row) * 64 + cch) * Sn + (t + 1) + wj;
        *bp += u;
      }
    }
    __threadfence();
    __syncthreads();
  }
}

// ---------------- prediction head ----------------
__global__ __launch_bounds__(256) void k_pred(
    const float* __restrict__ gfs, const _Float16* __restrict__ WP1, const float* __restrict__ BP1,
    const _Float16* __restrict__ WP2, const float* __restrict__ BP2, float* __restrict__ out) {
  extern __shared__ char smem[];
  _Float16* sA = (_Float16*)smem;                  // 64*256
  _Float16* sP = sA + 64 * 256;                    // 64*512
  int row0 = blockIdx.x * 64;
  for (int i = threadIdx.x; i < 64 * 256; i += 256)
    sA[i] = (_Float16)gfs[(size_t)row0 * 256 + i];
  __syncthreads();
  int wave = threadIdx.x >> 5, nl = laneid() & 15, hl = laneid() >> 4;
  {
    int m0 = (wave >> 1) * 16, nb = (wave & 1) * 16;
    v8f acc[16];
#pragma unroll
    for (int q = 0; q < 16; ++q) acc[q] = {};
    v16h bf = ld_b(WP1, nb * 16, 256, 0);
    for (int ks = 0; ks < 8; ++ks) {
      v16h a = ld_a(sA + m0 * 256 + ks * 32, 256);
#pragma unroll
      for (int q = 0; q < 16; ++q) {
        v16h bn = ld_b(WP1, (nb + ((q + 1) & 15)) * 16, 256, (ks + (q == 15)) * 32);
        acc[q] = wmma_f16(a, bf, acc[q]);
        bf = bn;
      }
    }
#pragma unroll
    for (int q = 0; q < 16; ++q) {
      int col = (nb + q) * 16 + nl;
      float bb = BP1[col];
#pragma unroll
      for (int r = 0; r < 8; ++r)
        sP[(m0 + r + 8 * hl) * 512 + col] = (_Float16)fmaxf(acc[q][r] + bb, 0.f);
    }
  }
  __syncthreads();
  for (int i = threadIdx.x; i < 64 * 4; i += 256) {
    int r = i >> 2, o = i & 3;
    float a = BP2[o];
    const _Float16* Wp = WP2 + o * 512;
    for (int k = 0; k < 512; ++k) a += (float)Wp[k] * (float)sP[r * 512 + k];
    out[(size_t)(row0 + r) * 4 + o] = fmaxf(a, 0.f);
  }
}

// ---------------- host launcher ----------------
extern "C" void kernel_launch(void* const* d_in, const int* in_sizes, int n_in,
                              void* d_out, int out_size, void* d_ws, size_t ws_size,
                              hipStream_t stream) {
  (void)in_sizes; (void)n_in; (void)out_size; (void)ws_size;
  const float* reads = (const float*)d_in[0];
  const float* w_emb[3] = {(const float*)d_in[1], (const float*)d_in[3], (const float*)d_in[5]};
  const float* b_emb[3] = {(const float*)d_in[2], (const float*)d_in[4], (const float*)d_in[6]};
  const float* w_bel[3] = {(const float*)d_in[7], (const float*)d_in[9], (const float*)d_in[11]};
  const float* b_bel[3] = {(const float*)d_in[8], (const float*)d_in[10], (const float*)d_in[12]};
  const float* w1  = (const float*)d_in[13]; const float* b1  = (const float*)d_in[14];
  const float* w2  = (const float*)d_in[15]; const float* b2  = (const float*)d_in[16];
  const float* wih = (const float*)d_in[17]; const float* bih = (const float*)d_in[18];
  const float* whh = (const float*)d_in[19]; const float* bhh = (const float*)d_in[20];
  const float* wq  = (const float*)d_in[21]; const float* bq  = (const float*)d_in[22];
  const float* wk  = (const float*)d_in[23]; const float* bk  = (const float*)d_in[24];
  const float* wv  = (const float*)d_in[25]; const float* bv  = (const float*)d_in[26];
  const float* wg  = (const float*)d_in[27]; const float* bg  = (const float*)d_in[28];
  const float* wu1 = (const float*)d_in[29]; const float* bu1 = (const float*)d_in[30];
  const float* wu2 = (const float*)d_in[31]; const float* bu2 = (const float*)d_in[32];
  const float* wp1 = (const float*)d_in[33]; const float* bp1 = (const float*)d_in[34];
  const float* wp2 = (const float*)d_in[35]; const float* bp2 = (const float*)d_in[36];

  char* ws = (char*)d_ws;
  size_t off = 0;
  auto alloc = [&](size_t bytes) { size_t o = off; off = (off + bytes + 255) & ~(size_t)255; return o; };
  _Float16* xin     = (_Float16*)(ws + alloc((size_t)Nn * 5 * Sn * 2));
  _Float16* tmpA    = (_Float16*)(ws + alloc((size_t)Nn * Cn * Sn * 2));
  _Float16* tmpB    = (_Float16*)(ws + alloc((size_t)Nn * Cn * Sn * 2));
  float*    beliefB = (float*)   (ws + alloc((size_t)Nn * Cn * Sn * 4));
  float*    gfs     = (float*)   (ws + alloc((size_t)Bn * Tn * Dn * 4));
  int kpads[3] = {32, 192, 192};
  _Float16* cw_emb[3]; _Float16* cw_bel[3];
  for (int i = 0; i < 3; ++i) cw_emb[i] = (_Float16*)(ws + alloc(64 * kpads[i] * 2));
  for (int i = 0; i < 3; ++i) cw_bel[i] = (_Float16*)(ws + alloc(64 * kpads[i] * 2));
  _Float16* f_w1  = (_Float16*)(ws + alloc(512 * 1024 * 2));
  _Float16* f_w2  = (_Float16*)(ws + alloc(256 * 512 * 2));
  _Float16* f_wl  = (_Float16*)(ws + alloc(1024 * 512 * 2));
  _Float16* f_wq  = (_Float16*)(ws + alloc(4 * 256 * 256 * 2));
  _Float16* f_wk  = (_Float16*)(ws + alloc(4 * 256 * 256 * 2));
  _Float16* f_wv  = (_Float16*)(ws + alloc(4 * 256 * 256 * 2));
  _Float16* f_wg  = (_Float16*)(ws + alloc(256 * 1024 * 2));
  _Float16* f_wu1 = (_Float16*)(ws + alloc(512 * 512 * 2));
  _Float16* f_wu2 = (_Float16*)(ws + alloc(448 * 512 * 2));
  _Float16* f_wp1 = (_Float16*)(ws + alloc(512 * 256 * 2));
  _Float16* f_wp2 = (_Float16*)(ws + alloc(4 * 512 * 2 + 256)); // +pad for pipelined over-read

  for (int i = 0; i < 3; ++i) {
    int cin = (i == 0) ? 5 : 64;
    int tot = 64 * kpads[i];
    k_cvt_conv<<<(tot + 255) / 256, 256, 0, stream>>>(w_emb[i], cw_emb[i], cin, kpads[i]);
    k_cvt_conv<<<(tot + 255) / 256, 256, 0, stream>>>(w_bel[i], cw_bel[i], cin, kpads[i]);
  }
  auto CVT = [&](const float* s, _Float16* d, int n) {
    k_cvt<<<(n + 255) / 256, 256, 0, stream>>>(s, d, n);
  };
  CVT(w1, f_w1, 512 * 1024);  CVT(w2, f_w2, 256 * 512);
  k_cvt_lstm<<<(1024 * 512 + 255) / 256, 256, 0, stream>>>(wih, whh, f_wl);
  CVT(wq, f_wq, 4 * 256 * 256); CVT(wk, f_wk, 4 * 256 * 256); CVT(wv, f_wv, 4 * 256 * 256);
  CVT(wg, f_wg, 256 * 1024);  CVT(wu1, f_wu1, 512 * 512); CVT(wu2, f_wu2, 448 * 512);
  CVT(wp1, f_wp1, 512 * 256); CVT(wp2, f_wp2, 4 * 512);

  k_transpose<<<Nn, 256, 0, stream>>>(reads, xin);

  dim3 cgrid(Sn / 128, Nn);
  auto csm = [](int cin, int kpad) {
    return (size_t)(128 * (kpad + 16) + cin * 132) * 2;
  };
  // belief initializer convnet -> f32 belief buffer
  k_conv<false><<<cgrid, 256, csm(5, 32), stream>>>(xin, cw_bel[0], b_bel[0], tmpA, 5, 32);
  k_conv<false><<<cgrid, 256, csm(64, 192), stream>>>(tmpA, cw_bel[1], b_bel[1], tmpB, 64, 192);
  k_conv<true ><<<cgrid, 256, csm(64, 192), stream>>>(tmpB, cw_bel[2], b_bel[2], beliefB, 64, 192);
  // embedding convnet -> f16 original buffer (tmpA)
  k_conv<false><<<cgrid, 256, csm(5, 32), stream>>>(xin, cw_emb[0], b_emb[0], tmpA, 5, 32);
  k_conv<false><<<cgrid, 256, csm(64, 192), stream>>>(tmpA, cw_emb[1], b_emb[1], tmpB, 64, 192);
  k_conv<false><<<cgrid, 256, csm(64, 192), stream>>>(tmpB, cw_emb[2], b_emb[2], tmpA, 64, 192);

  size_t smem_scan = (size_t)(64 * 512 * 2) * 3 + 64 * 256 * 4 + 64 * 256 * 2 +
                     64 * 64 * 4 + (64 + 256 + 1024 + 256) * 4;
  k_scan<<<Bn, 512, smem_scan, stream>>>(tmpA, beliefB,
      f_w1, b1, f_w2, b2, f_wl, bih, bhh,
      f_wq, bq, f_wk, bk, f_wv, bv, f_wg, bg,
      f_wu1, bu1, f_wu2, bu2, gfs);

  size_t smem_pred = 64 * 256 * 2 + 64 * 512 * 2;
  k_pred<<<(Bn * Tn) / 64, 256, smem_pred, stream>>>(gfs, f_wp1, bp1, f_wp2, bp2, (float*)d_out);
}
